// RWKV6TimeMix_10359461117970
// MI455X (gfx1250) — matrix operand order repaired and checked by reading, not compiled, not run on hardware
//
#include <hip/hip_runtime.h>
#include <hip/hip_bf16.h>

// ---------------------------------------------------------------------------
// RWKV6 TimeMix for MI455X (gfx1250): f16 WMMA GEMM pipeline + VALU WKV scan
// GEMM uses CDNA5 async global->LDS copies (ASYNCcnt) with double-buffered
// LDS tiles; all LDS traffic is b128; weights pre-transposed so A and B
// fragments are contiguous 16B runs matching the ISA WMMA register layouts.
// ---------------------------------------------------------------------------

typedef __attribute__((ext_vector_type(16))) _Float16     v16h;
typedef __attribute__((ext_vector_type(8)))  float        v8f;
typedef __attribute__((ext_vector_type(4)))  unsigned int v4u;

union Frag16 { v16h h; v4u u[2]; };

#define RB 4
#define RT 2048
#define RC 2048
#define RH 32
#define RHS 64
#define LN_EPS 6.4e-4f   // 1e-5 * 8^2

enum { EPI_NONE = 0, EPI_TANH, EPI_SILU, EPI_DECAY, EPI_MIX };

// CDNA5 async copy: global memory -> LDS, 16B per lane, tracked by ASYNCcnt
__device__ __forceinline__ void async_b128(unsigned lds_off, const _Float16* g) {
    asm volatile("global_load_async_to_lds_b128 %0, %1, off"
                 :: "v"(lds_off), "v"(g) : "memory");
}
__device__ __forceinline__ void wait_async0() {
    asm volatile("s_wait_asynccnt 0x0" ::: "memory");
}

// ---------------------------------------------------------------------------
// f32 -> f16 transpose-convert:  d[n*K + k] = (f16) s[k*N + n]
// (coalesced reads; makes B^T row-major so GEMM staging is all-b128)
// ---------------------------------------------------------------------------
__global__ void cvt_f32_f16_t(const float* __restrict__ s, _Float16* __restrict__ d,
                              int K, int N) {
    size_t i = (size_t)blockIdx.x * 256 + threadIdx.x;
    if (i >= (size_t)K * N) return;
    int n = (int)(i % N), k = (int)(i / N);
    d[(size_t)n * K + k] = (_Float16)s[i];
}

// ---------------------------------------------------------------------------
// Token shift prep: dxprev = shift(x) - x ; mixed = x + dxprev*maa_x (f16)
// also writes shift_state_out
// ---------------------------------------------------------------------------
__global__ void prep_kernel(const float* __restrict__ x,
                            const float* __restrict__ shift_in,
                            const float* __restrict__ maa_x,
                            float* __restrict__ dxprev,
                            _Float16* __restrict__ mixed,
                            float* __restrict__ shift_out) {
    size_t idx = (size_t)blockIdx.x * 256 + threadIdx.x;
    int c = (int)(idx % RC);
    int t = (int)((idx / RC) % RT);
    int b = (int)(idx / ((size_t)RC * RT));
    float xv = x[idx];
    float prev = (t == 0) ? shift_in[(size_t)b * RC + c] : x[idx - RC];
    float dx = prev - xv;
    dxprev[idx] = dx;
    mixed[idx] = (_Float16)(xv + dx * maa_x[c]);
    if (t == RT - 1) shift_out[(size_t)b * RC + c] = xv;
}

// ---------------------------------------------------------------------------
// Tiled f16 WMMA GEMM:  Out[M,N] = epilogue( A[M,K] @ B[K,N] )
// B supplied TRANSPOSED row-major: Bt[n*K + k].
// Block: 256 threads (8 waves, 4Mx2N wave grid). Tile: 128(M) x 128(N),
// K step 32, double-buffered LDS filled by async global->LDS b128 copies.
// Each wave: 2 M-frags x 4 N-frags = 8 WMMA per K-step.
// Requires: M % 128 == 0, K % 32 == 0, lda/K multiples of 8 halves (N guarded).
// ---------------------------------------------------------------------------
template<int EPI, bool OUTF16>
__global__ __launch_bounds__(256)
void gemm_f16_wmma(const _Float16* __restrict__ A, int lda,
                   const _Float16* __restrict__ Bt,
                   void* __restrict__ Out, int ldc,
                   int M, int N, int K,
                   const float* __restrict__ ep_vec,
                   const float* __restrict__ ep_x,
                   const float* __restrict__ ep_dx) {
    // 40-half row stride: rows stay 16B aligned (80B = 5*16B), bank-conflict-free
    __shared__ __align__(16) _Float16 As [2][128][40];
    __shared__ __align__(16) _Float16 Bst[2][128][40];   // B^T tile: [n][k]

    const int tid   = threadIdx.x;
    const int lane  = tid & 31;
    const int wv    = tid >> 5;
    const int wv_m  = wv >> 1;          // 0..3 -> M sub-tile of 32 rows
    const int wv_n  = wv & 1;           // 0..1 -> N sub-tile of 64 cols
    const int khalf = lane >> 4;
    const int l16   = lane & 15;
    const int m0    = blockIdx.y * 128;
    const int n0    = blockIdx.x * 128;

    // zero OOB B^T rows once (row validity depends only on n, not k)
    if (n0 + 128 > N) {
        for (int c = tid; c < 512; c += 256) {
            int r = c >> 2, koff = (c & 3) * 8;
            if (n0 + r >= N) {
                v4u z = {0u, 0u, 0u, 0u};
                *(v4u*)&Bst[0][r][koff] = z;
                *(v4u*)&Bst[1][r][koff] = z;
            }
        }
    }

    const int nsteps = K >> 5;

    auto issue = [&](int step) {
        const int buf = step & 1;
        const int k0  = step << 5;
        #pragma unroll
        for (int c = tid; c < 512; c += 256) {           // A: 128x32 halves
            int r = c >> 2, koff = (c & 3) * 8;
            async_b128((unsigned)(size_t)&As[buf][r][koff],
                       &A[(size_t)(m0 + r) * lda + (k0 + koff)]);
        }
        #pragma unroll
        for (int c = tid; c < 512; c += 256) {           // B^T: 128x32 halves
            int r = c >> 2, koff = (c & 3) * 8;
            if (n0 + r < N)
                async_b128((unsigned)(size_t)&Bst[buf][r][koff],
                           &Bt[(size_t)(n0 + r) * K + (k0 + koff)]);
        }
    };

    v8f acc[2][4] = {};

    issue(0);
    for (int step = 0; step < nsteps; ++step) {
        const int buf = step & 1;
        wait_async0();        // my tile-`step` copies have landed in LDS
        __syncthreads();      // everyone's copies landed; other buffer free
        if (step + 1 < nsteps) issue(step + 1);   // overlap with compute below

        // A fragments (ISA 16-bit A 16x32): lane holds K khalf*8..+7, 16+khalf*8..+7
        Frag16 af[2];
        #pragma unroll
        for (int im = 0; im < 2; ++im) {
            const int ra = wv_m * 32 + im * 16 + l16;
            af[im].u[0] = *(const v4u*)&As[buf][ra][khalf * 8];
            af[im].u[1] = *(const v4u*)&As[buf][ra][16 + khalf * 8];
        }
        #pragma unroll
        for (int j = 0; j < 4; ++j) {
            // B fragment: lane = column, K khalf*8..+7, 16+khalf*8..+7 (same as A)
            Frag16 bf;
            const int cb = wv_n * 64 + j * 16 + l16;
            bf.u[0] = *(const v4u*)&Bst[buf][cb][khalf * 8];
            bf.u[1] = *(const v4u*)&Bst[buf][cb][16 + khalf * 8];
            #pragma unroll
            for (int im = 0; im < 2; ++im)
                acc[im][j] = __builtin_amdgcn_wmma_f32_16x16x32_f16(
                    false, af[im].h, false, bf.h, (short)0, acc[im][j], false, false);
        }
    }

    // store with fused epilogue.  D layout: VGPR e -> row khalf*8+e, col = l16
    #pragma unroll
    for (int im = 0; im < 2; ++im) {
        #pragma unroll
        for (int j = 0; j < 4; ++j) {
            int col = n0 + wv_n * 64 + j * 16 + l16;
            if (col >= N) continue;
            #pragma unroll
            for (int e = 0; e < 8; ++e) {
                int row = m0 + wv_m * 32 + im * 16 + khalf * 8 + e;
                size_t oidx = (size_t)row * ldc + col;
                float val = acc[im][j][e];
                if (EPI == EPI_TANH)  val = tanhf(val);
                if (EPI == EPI_SILU)  val = val / (1.f + expf(-val));
                if (EPI == EPI_DECAY) val = expf(-expf(ep_vec[col] + val));
                if (EPI == EPI_MIX)   val = ep_x[oidx] + ep_dx[oidx] * (ep_vec[col] + val);
                if (OUTF16) ((_Float16*)Out)[oidx] = (_Float16)val;
                else        ((float*)Out)[oidx]    = val;
            }
        }
    }
}

// ---------------------------------------------------------------------------
// WKV scan: one block per (b,h). 256 threads: thread = (n = tid>>2, kq = tid&3)
// owns S[kq*16+i][n]. Per step: LDS-broadcast r/k/w, butterfly-reduce y over kq.
// ---------------------------------------------------------------------------
__global__ __launch_bounds__(256)
void wkv_scan_kernel(const float* __restrict__ rr, const float* __restrict__ kk_,
                     const float* __restrict__ vv, const float* __restrict__ wd,
                     const float* __restrict__ faaaa,
                     const float* __restrict__ s_in,
                     float* __restrict__ y, float* __restrict__ s_out) {
    const int bh = blockIdx.x;
    const int b = bh / RH, h = bh % RH;
    const int tid = threadIdx.x;
    const int n = tid >> 2, kq = tid & 3;

    __shared__ float rs[64], ks[64], wds[64], us[64];
    if (tid < 64) us[tid] = faaaa[h * 64 + tid];

    float S[16];
    #pragma unroll
    for (int i = 0; i < 16; ++i)
        S[i] = s_in[((size_t)(b * RH + h) * 64 + (kq * 16 + i)) * 64 + n];

    for (int t = 0; t < RT; ++t) {
        size_t off = ((size_t)b * RT + t) * RC + h * 64;
        if (tid < 64)       rs[tid]        = rr[off + tid];
        else if (tid < 128) ks[tid - 64]   = kk_[off + tid - 64];
        else if (tid < 192) wds[tid - 128] = wd[off + tid - 128];
        __syncthreads();

        float vn = vv[off + n];
        float yp = 0.f;
        #pragma unroll
        for (int i = 0; i < 16; ++i) {
            int kk = kq * 16 + i;
            float kv = ks[kk] * vn;
            yp += rs[kk] * fmaf(us[kk], kv, S[i]);
            S[i] = fmaf(wds[kk], S[i], kv);
        }
        // reduce over kq (adjacent lanes in wave32)
        yp += __shfl_xor(yp, 1, 32);
        yp += __shfl_xor(yp, 2, 32);
        if (kq == 0) y[off + n] = yp;
        __syncthreads();
    }

    #pragma unroll
    for (int i = 0; i < 16; ++i)
        s_out[((size_t)(b * RH + h) * 64 + (kq * 16 + i)) * 64 + n] = S[i];
}

// ---------------------------------------------------------------------------
// Fused per-head LayerNorm(HS=64) * gamma + beta, times g -> f16 for final GEMM
// One wave per head; lane handles elements {lane, lane+32}.
// ---------------------------------------------------------------------------
__global__ __launch_bounds__(256)
void ln_g_kernel(const float* __restrict__ y, const float* __restrict__ g,
                 const float* __restrict__ gamma, const float* __restrict__ beta,
                 _Float16* __restrict__ z) {
    int wid  = blockIdx.x * 8 + (threadIdx.x >> 5); // (b*T+t)*H + h
    int lane = threadIdx.x & 31;
    int h    = wid & (RH - 1);
    size_t row  = (size_t)(wid >> 5);               // b*T + t  (H == 32)
    size_t base = row * RC + (size_t)h * 64;

    float y0 = y[base + lane], y1 = y[base + 32 + lane];
    float s = y0 + y1;
    #pragma unroll
    for (int m = 16; m >= 1; m >>= 1) s += __shfl_xor(s, m, 32);
    float mu = s * (1.f / 64.f);
    float d0 = y0 - mu, d1 = y1 - mu;
    float q = d0 * d0 + d1 * d1;
    #pragma unroll
    for (int m = 16; m >= 1; m >>= 1) q += __shfl_xor(q, m, 32);
    float rstd = rsqrtf(q * (1.f / 64.f) + LN_EPS);

    int c0 = h * 64 + lane, c1 = c0 + 32;
    z[base + lane]      = (_Float16)((d0 * rstd * gamma[c0] + beta[c0]) * g[base + lane]);
    z[base + 32 + lane] = (_Float16)((d1 * rstd * gamma[c1] + beta[c1]) * g[base + 32 + lane]);
}

// ---------------------------------------------------------------------------
// Host launcher
// ---------------------------------------------------------------------------
extern "C" void kernel_launch(void* const* d_in, const int* in_sizes, int n_in,
                              void* d_out, int out_size, void* d_ws, size_t ws_size,
                              hipStream_t stream) {
    (void)in_sizes; (void)n_in; (void)out_size; (void)ws_size;

    const float* x        = (const float*)d_in[0];
    const float* shift_in = (const float*)d_in[1];
    const float* wkv_in   = (const float*)d_in[2];
    const float* maa_x    = (const float*)d_in[3];
    const float* maa_5[5] = { (const float*)d_in[4], (const float*)d_in[5],
                              (const float*)d_in[6], (const float*)d_in[7],
                              (const float*)d_in[8] };           // w,k,v,r,g
    const float* maa_w1   = (const float*)d_in[9];
    const float* maa_w2   = (const float*)d_in[10];
    const float* tdecay   = (const float*)d_in[11];
    const float* dw1      = (const float*)d_in[12];
    const float* dw2      = (const float*)d_in[13];
    const float* faaaa    = (const float*)d_in[14];
    const float* Wbig[5]  = { (const float*)d_in[15], (const float*)d_in[16],
                              (const float*)d_in[17], (const float*)d_in[18],
                              (const float*)d_in[19] };          // r,k,v,g,o
    const float* gamma    = (const float*)d_in[20];
    const float* beta     = (const float*)d_in[21];

    const size_t BT = (size_t)RB * RT;                           // 8192
    float* out_f     = (float*)d_out;
    float* shift_out = out_f + BT * RC;
    float* wkv_out   = shift_out + (size_t)RB * RC;

    // ---- workspace layout ----
    char* ws = (char*)d_ws;
    size_t o = 0;
    auto take = [&](size_t bytes) { size_t r = o; o += (bytes + 255) & ~(size_t)255; return r; };
    size_t o_w16[5]; for (int i = 0; i < 5; ++i) o_w16[i] = take((size_t)RC * RC * 2);
    size_t o_w1_16  = take((size_t)RC * 160 * 2);        // (160 x 2048) transposed
    size_t o_w2_16  = take((size_t)5 * 32 * RC * 2);     // 5 x (2048 x 32) transposed
    size_t o_dw1_16 = take((size_t)RC * 64 * 2);         // (64 x 2048) transposed
    size_t o_dw2_16 = take((size_t)64 * RC * 2);         // (2048 x 64) transposed
    size_t o_dxprev = take(BT * RC * 4);      // later reused for g (f32)
    size_t o_mixed  = take(BT * RC * 2);      // later reused for z (f16)
    size_t o_xxx    = take(BT * 160 * 2);     // later reused for t (f16, BT*64)
    size_t o_x5     = take((size_t)5 * BT * RC * 2); // xw,xk,xv,xr,xg
    size_t o_r      = take(BT * RC * 4);
    size_t o_k      = take(BT * RC * 4);
    size_t o_v      = take(BT * RC * 4);
    size_t o_decay  = take(BT * RC * 4);
    size_t o_g      = o_dxprev;               // alias (dxprev dead after mix GEMMs)
    size_t o_y      = o_x5;                   // alias (xw+xk dead after decay/k GEMMs)
    size_t o_z      = o_mixed;                // alias (mixed dead after xxx GEMM)
    size_t o_t16    = o_xxx;                  // alias (xxx dead after mix GEMMs)

    _Float16* w16[5]; for (int i = 0; i < 5; ++i) w16[i] = (_Float16*)(ws + o_w16[i]);
    _Float16* w1_16  = (_Float16*)(ws + o_w1_16);
    _Float16* w2_16  = (_Float16*)(ws + o_w2_16);
    _Float16* dw1_16 = (_Float16*)(ws + o_dw1_16);
    _Float16* dw2_16 = (_Float16*)(ws + o_dw2_16);
    float*    dxprev = (float*)(ws + o_dxprev);
    _Float16* mixed  = (_Float16*)(ws + o_mixed);
    _Float16* xxx16  = (_Float16*)(ws + o_xxx);
    _Float16* x5_16  = (_Float16*)(ws + o_x5);
    float*    r_f    = (float*)(ws + o_r);
    float*    k_f    = (float*)(ws + o_k);
    float*    v_f    = (float*)(ws + o_v);
    float*    dec_f  = (float*)(ws + o_decay);
    float*    g_f    = (float*)(ws + o_g);
    float*    y_f    = (float*)(ws + o_y);
    _Float16* z16    = (_Float16*)(ws + o_z);
    _Float16* t16    = (_Float16*)(ws + o_t16);

    auto cvt_t = [&](const float* src, _Float16* dst, int K, int N) {
        size_t n = (size_t)K * N;
        cvt_f32_f16_t<<<(unsigned)((n + 255) / 256), 256, 0, stream>>>(src, dst, K, N);
    };

    // 1) weight conversions to f16, transposed (B^T row-major)
    for (int i = 0; i < 5; ++i) cvt_t(Wbig[i], w16[i], RC, RC);
    cvt_t(maa_w1, w1_16, RC, 160);
    for (int c = 0; c < 5; ++c)
        cvt_t(maa_w2 + (size_t)c * 32 * RC, w2_16 + (size_t)c * RC * 32, 32, RC);
    cvt_t(dw1, dw1_16, RC, 64);
    cvt_t(dw2, dw2_16, 64, RC);

    // 2) token shift prep
    prep_kernel<<<(unsigned)(BT * RC / 256), 256, 0, stream>>>(
        x, shift_in, maa_x, dxprev, mixed, shift_out);

    dim3 blk(256);
    // 3) xxx = tanh(mixed @ maa_w1)   [8192 x 2048 x 160] -> f16
    {
        dim3 g((160 + 127) / 128, (unsigned)(BT / 128));
        gemm_f16_wmma<EPI_TANH, true><<<g, blk, 0, stream>>>(
            mixed, RC, w1_16, (void*)xxx16, 160,
            (int)BT, 160, RC, nullptr, nullptr, nullptr);
    }
    // 4) five mix GEMMs: x* = x + dxprev*(maa_* + xxx_chunk @ w2_chunk) -> f16
    for (int c = 0; c < 5; ++c) {
        dim3 g(RC / 128, (unsigned)(BT / 128));
        gemm_f16_wmma<EPI_MIX, true><<<g, blk, 0, stream>>>(
            xxx16 + c * 32, 160, w2_16 + (size_t)c * RC * 32,
            (void*)(x5_16 + (size_t)c * BT * RC), RC,
            (int)BT, RC, 32, maa_5[c], x, dxprev);
    }
    // 5) projections: r = xr@Wr, k = xk@Wk, v = xv@Wv (f32), g = silu(xg@Wg)
    {
        dim3 g(RC / 128, (unsigned)(BT / 128));
        gemm_f16_wmma<EPI_NONE, false><<<g, blk, 0, stream>>>(
            x5_16 + (size_t)3 * BT * RC, RC, w16[0], (void*)r_f, RC,
            (int)BT, RC, RC, nullptr, nullptr, nullptr);
        gemm_f16_wmma<EPI_NONE, false><<<g, blk, 0, stream>>>(
            x5_16 + (size_t)1 * BT * RC, RC, w16[1], (void*)k_f, RC,
            (int)BT, RC, RC, nullptr, nullptr, nullptr);
        gemm_f16_wmma<EPI_NONE, false><<<g, blk, 0, stream>>>(
            x5_16 + (size_t)2 * BT * RC, RC, w16[2], (void*)v_f, RC,
            (int)BT, RC, RC, nullptr, nullptr, nullptr);
        gemm_f16_wmma<EPI_SILU, false><<<g, blk, 0, stream>>>(
            x5_16 + (size_t)4 * BT * RC, RC, w16[3], (void*)g_f, RC,
            (int)BT, RC, RC, nullptr, nullptr, nullptr);
    }
    // 6) decay path: t = tanh(xw @ dw1) (f16); decay = exp(-exp(tdecay + t@dw2))
    {
        dim3 g1(1, (unsigned)(BT / 128));
        gemm_f16_wmma<EPI_TANH, true><<<g1, blk, 0, stream>>>(
            x5_16, RC, dw1_16, (void*)t16, 64,
            (int)BT, 64, RC, nullptr, nullptr, nullptr);
        dim3 g2(RC / 128, (unsigned)(BT / 128));
        gemm_f16_wmma<EPI_DECAY, false><<<g2, blk, 0, stream>>>(
            t16, 64, dw2_16, (void*)dec_f, RC,
            (int)BT, RC, 64, tdecay, nullptr, nullptr);
    }
    // 7) WKV recurrence
    wkv_scan_kernel<<<RB * RH, 256, 0, stream>>>(
        r_f, k_f, v_f, dec_f, faaaa, wkv_in, y_f, wkv_out);
    // 8) per-head LayerNorm * g -> z (f16)
    ln_g_kernel<<<(unsigned)(BT * RH / 8), 256, 0, stream>>>(
        y_f, g_f, gamma, beta, z16);
    // 9) out = z @ Wo (f32 into d_out)
    {
        dim3 g(RC / 128, (unsigned)(BT / 128));
        gemm_f16_wmma<EPI_NONE, false><<<g, blk, 0, stream>>>(
            z16, RC, w16[4], (void*)out_f, RC,
            (int)BT, RC, RC, nullptr, nullptr, nullptr);
    }
}
